// ROIHeadPostprocess_12283606468109
// MI455X (gfx1250) — compile-verified, matrix operand order checked by reading, not run on hardware
//
#include <hip/hip_runtime.h>
#include <hip/hip_bf16.h>
#include <cfloat>
#include <cmath>

// ---------------------------------------------------------------------------
// ROI head postprocess for MI455X (gfx1250, wave32).
// Kernel 1: per-ROI softmax-max/argmax + box decode. The softmax denominator
//           (row-sum of exp) is computed with v_wmma_f32_16x16x32_f16:
//           16 ROIs per wave, B-matrix = all-ones, f32 accumulate.
// Kernel 2: per-image batched NMS (bitonic sort + greedy suppression in LDS).
// ---------------------------------------------------------------------------

typedef __attribute__((ext_vector_type(16))) _Float16 v16h;
typedef __attribute__((ext_vector_type(8)))  float    v8f;

#define RB 16      // batch
#define RN 1024    // rois per image
#define RC 91      // classes
#define MAXDET 100
#define SCORE_THRESH 0.05f
#define NMS_THRESH 0.5f
#define XFORM_CLIP 4.135166556742356f   // log(1000/16)

// ---------------------------------------------------------------------------
// Kernel 1: one wave32 handles 16 consecutive ROIs.
//   lane L: m = L&15 (ROI within the 16), g = L>>4 (class-group half).
//   Classes covered by lane over 3 WMMA steps: k = step*32 + g*16 + [0..15].
//   Only (g==1, step==2, j>=11) maps out of range (k in 91..95): those lanes
//   re-read k-16 (always valid) and zero the contribution via select —
//   no divergent EXEC-mask blocks, loads stay vectorizable.
// ---------------------------------------------------------------------------
__global__ __launch_bounds__(256)
void roi_decode_kernel(const float* __restrict__ logits,
                       const float* __restrict__ deltas,
                       const float* __restrict__ props,
                       const int*   __restrict__ sizes,
                       float* __restrict__ wsBoxes,
                       float* __restrict__ wsScores,
                       int*   __restrict__ wsLabels,
                       int*   __restrict__ wsValid) {
    const int gtid = blockIdx.x * blockDim.x + threadIdx.x;
    const int wave = gtid >> 5;
    const int lane = threadIdx.x & 31;
    const int m    = lane & 15;
    const int g    = lane >> 4;
    const int roi  = wave * 16 + m;           // < 16384 by grid construction

    const float* row = logits + (size_t)roi * RC;
    __builtin_prefetch(row, 0, 0);            // global_prefetch_b8

    // ---- load this lane's 48 class logits once (branch-free tail) ---------
    float r[3][16];
    bool  ok[3][16];
    #pragma unroll
    for (int step = 0; step < 3; ++step) {
        #pragma unroll
        for (int j = 0; j < 16; ++j) {
            int  k   = step * 32 + g * 16 + j;
            bool val = (step < 2) || (j < 11) || (g == 0);   // k < 91 ?
            int  kk  = val ? k : (k - 16);                   // safe address
            r[step][j]  = row[kk];
            ok[step][j] = val;
        }
    }

    // ---- pass 1: max / argmax over this lane's classes --------------------
    float mx = -FLT_MAX;
    int   arg = 0;
    #pragma unroll
    for (int step = 0; step < 3; ++step) {
        #pragma unroll
        for (int j = 0; j < 16; ++j) {
            int   k = step * 32 + g * 16 + j;
            float v = r[step][j];
            if (ok[step][j] && v > mx) { mx = v; arg = k; }
        }
    }
    // combine with the partner lane (same ROI, other class half).
    {
        float omx  = __shfl_xor(mx,  16);
        int   oarg = __shfl_xor(arg, 16);
        if (omx > mx || (omx == mx && oarg < arg)) { mx = omx; arg = oarg; }
    }

    // ---- pass 2: sumexp via WMMA (B = ones => row sums) -------------------
    v16h ones;
    #pragma unroll
    for (int j = 0; j < 16; ++j) ones[j] = (_Float16)1.0f;

    v8f acc = {};
    #pragma unroll
    for (int step = 0; step < 3; ++step) {
        v16h a;
        #pragma unroll
        for (int j = 0; j < 16; ++j) {
            float e = __expf(r[step][j] - mx);
            e = ok[step][j] ? e : 0.0f;                      // v_cndmask, no branch
            a[j] = (_Float16)e;
        }
        acc = __builtin_amdgcn_wmma_f32_16x16x32_f16(
            /*neg_a=*/false, a, /*neg_b=*/false, ones,
            /*c_mod=*/(short)0, acc, /*reuse_a=*/false, /*reuse_b=*/false);
    }

    // D layout (f32 16x16 C/D): lane 0 holds M=0..7 in v0..v7 (N=0),
    // lane 16 holds M=8..15 in v0..v7. Broadcast row sums to lanes 0..15.
    float sumexp = 1.0f;
    #pragma unroll
    for (int mm = 0; mm < 8; ++mm) {
        float t0 = __shfl(acc[mm], 0);
        float t1 = __shfl(acc[mm], 16);
        if (lane == mm)     sumexp = t0;
        if (lane == mm + 8) sumexp = t1;
    }

    // ---- finalize (lanes 0..15 only; one ROI each) ------------------------
    if (g == 0) {
        const float score = 1.0f / sumexp;        // exp(0)/sumexp
        const int   label = arg;
        const int   valid = (label > 0) && (score > SCORE_THRESH);

        const int b = roi >> 10;
        const float* p = props  + (size_t)roi * 4;
        const float* d = deltas + (size_t)roi * (RC * 4) + (size_t)label * 4;

        float w  = p[2] - p[0];
        float h  = p[3] - p[1];
        float cx = p[0] + 0.5f * w;
        float cy = p[1] + 0.5f * h;
        float dx = d[0], dy = d[1];
        float dw = fminf(d[2], XFORM_CLIP);
        float dh = fminf(d[3], XFORM_CLIP);
        float pcx = dx * w + cx;
        float pcy = dy * h + cy;
        float pw  = __expf(dw) * w;
        float ph  = __expf(dh) * h;

        float x1 = pcx - 0.5f * pw, y1 = pcy - 0.5f * ph;
        float x2 = pcx + 0.5f * pw, y2 = pcy + 0.5f * ph;

        float H = (float)sizes[2 * b + 0];
        float W = (float)sizes[2 * b + 1];
        x1 = fminf(fmaxf(x1, 0.0f), W);
        y1 = fminf(fmaxf(y1, 0.0f), H);
        x2 = fminf(fmaxf(x2, 0.0f), W);
        y2 = fminf(fmaxf(y2, 0.0f), H);

        float* ob = wsBoxes + (size_t)roi * 4;
        ob[0] = x1; ob[1] = y1; ob[2] = x2; ob[3] = y2;
        wsScores[roi] = score;
        wsLabels[roi] = label;
        wsValid[roi]  = valid;
    }
}

// ---------------------------------------------------------------------------
// Kernel 2: per-image batched NMS. One block of 1024 threads per image.
// ---------------------------------------------------------------------------
__global__ __launch_bounds__(1024)
void roi_nms_kernel(const float* __restrict__ wsBoxes,
                    const float* __restrict__ wsScores,
                    const int*   __restrict__ wsLabels,
                    const int*   __restrict__ wsValid,
                    float* __restrict__ out) {
    __shared__ float  key[RN];
    __shared__ int    sidx[RN];
    __shared__ float4 obox[RN];
    __shared__ float  area[RN];
    __shared__ int    vs[RN];
    __shared__ int    sup[RN];
    __shared__ int    pfx[RN];
    __shared__ int    winner[MAXDET];
    __shared__ float  redbuf[32];
    __shared__ float  smax;

    const int b    = blockIdx.x;
    const int j    = threadIdx.x;
    const int base = b * RN;

    // load & key (invalid -> -FLT_MAX, matching jnp.finfo(f32).min behavior)
    {
        float s  = wsScores[base + j];
        int   vl = wsValid[base + j];
        key[j]  = vl ? s : -FLT_MAX;
        sidx[j] = j;
    }
    __syncthreads();

    // bitonic sort: (score desc, index asc) — emulates stable argsort(-s)
    for (int k = 2; k <= RN; k <<= 1) {
        for (int st = k >> 1; st > 0; st >>= 1) {
            __syncthreads();
            int p = j ^ st;
            if (p > j) {
                float ka = key[j], kb = key[p];
                int   ia = sidx[j], ib = sidx[p];
                bool aBeforeB = (ka > kb) || (ka == kb && ia < ib);
                bool desc = ((j & k) == 0);
                if (aBeforeB != desc) {
                    key[j] = kb; key[p] = ka;
                    sidx[j] = ib; sidx[p] = ia;
                }
            }
        }
    }
    __syncthreads();

    // gather per sorted slot
    const int oi = sidx[j];
    float4 bx;
    bx.x = wsBoxes[(size_t)(base + oi) * 4 + 0];
    bx.y = wsBoxes[(size_t)(base + oi) * 4 + 1];
    bx.z = wsBoxes[(size_t)(base + oi) * 4 + 2];
    bx.w = wsBoxes[(size_t)(base + oi) * 4 + 3];
    const int lb = wsLabels[base + oi];
    const int vl = wsValid[base + oi];

    // max coordinate over valid boxes (clipped boxes are >= 0, so init 0)
    {
        float mc = vl ? fmaxf(fmaxf(bx.x, bx.y), fmaxf(bx.z, bx.w)) : 0.0f;
        #pragma unroll
        for (int o = 16; o > 0; o >>= 1) mc = fmaxf(mc, __shfl_xor(mc, o));
        if ((j & 31) == 0) redbuf[j >> 5] = mc;
        __syncthreads();
        if (j < 32) {
            float t = redbuf[j];
            #pragma unroll
            for (int o = 16; o > 0; o >>= 1) t = fmaxf(t, __shfl_xor(t, o));
            if (j == 0) smax = t;
        }
        __syncthreads();
    }

    // class-offset boxes
    const float off = (float)lb * (smax + 1.0f);
    const float ox1 = bx.x + off, oy1 = bx.y + off;
    const float ox2 = bx.z + off, oy2 = bx.w + off;
    const float aj  = (ox2 - ox1) * (oy2 - oy1);
    obox[j] = make_float4(ox1, oy1, ox2, oy2);
    area[j] = aj;
    vs[j]   = vl;
    sup[j]  = 0;

    // greedy suppression (each thread owns sorted slot j)
    for (int i = 0; i < RN - 1; ++i) {
        __syncthreads();
        if (j > i && sup[i] == 0 && vs[i] != 0) {
            float4 bi = obox[i];
            float ix1 = fmaxf(bi.x, ox1);
            float iy1 = fmaxf(bi.y, oy1);
            float ix2 = fminf(bi.z, ox2);
            float iy2 = fminf(bi.w, oy2);
            float iw = fmaxf(ix2 - ix1, 0.0f);
            float ih = fmaxf(iy2 - iy1, 0.0f);
            float inter = iw * ih;
            float uni = area[i] + aj - inter;
            if (inter / (uni + 1e-9f) > NMS_THRESH) sup[j] = 1;
        }
    }
    __syncthreads();

    // compact kept entries (still score-descending) via prefix sum
    const int keep = (sup[j] == 0 && vl != 0) ? 1 : 0;
    pfx[j] = keep;
    for (int o = 1; o < RN; o <<= 1) {
        __syncthreads();
        int v = (j >= o) ? pfx[j - o] : 0;
        __syncthreads();
        pfx[j] += v;
    }
    __syncthreads();
    const int rank = pfx[j] - 1;

    if (j < MAXDET) winner[j] = -1;
    __syncthreads();
    if (keep && rank < MAXDET) winner[rank] = j;
    __syncthreads();

    // emit top-100; pad with boxes=0, scores=0, labels=-1
    if (j < MAXDET) {
        int wj = winner[j];
        float b0 = 0.f, b1 = 0.f, b2 = 0.f, b3 = 0.f;
        float sc = 0.f, lf = -1.f;
        if (wj >= 0) {
            int orig = base + sidx[wj];
            b0 = wsBoxes[(size_t)orig * 4 + 0];
            b1 = wsBoxes[(size_t)orig * 4 + 1];
            b2 = wsBoxes[(size_t)orig * 4 + 2];
            b3 = wsBoxes[(size_t)orig * 4 + 3];
            sc = key[wj];
            lf = (float)wsLabels[orig];
        }
        float* obx = out + ((size_t)b * MAXDET + j) * 4;     // boxes block
        obx[0] = b0; obx[1] = b1; obx[2] = b2; obx[3] = b3;
        out[RB * MAXDET * 4 + b * MAXDET + j]      = sc;     // scores block
        out[RB * MAXDET * 4 + RB * MAXDET + b * MAXDET + j] = lf; // labels
    }
}

// ---------------------------------------------------------------------------
extern "C" void kernel_launch(void* const* d_in, const int* in_sizes, int n_in,
                              void* d_out, int out_size, void* d_ws, size_t ws_size,
                              hipStream_t stream) {
    const float* logits = (const float*)d_in[0];   // [16,1024,91]
    const float* deltas = (const float*)d_in[1];   // [16,1024,364]
    const float* props  = (const float*)d_in[2];   // [16,1024,4]
    const int*   sizes  = (const int*)d_in[3];     // [16,2] (h,w)
    float* out = (float*)d_out;

    // workspace layout (all per-ROI, 16384 ROIs)
    float* wsBoxes  = (float*)d_ws;                      // 16384*4 f32
    float* wsScores = wsBoxes + (size_t)RB * RN * 4;     // 16384 f32
    int*   wsLabels = (int*)(wsScores + (size_t)RB * RN);// 16384 i32
    int*   wsValid  = wsLabels + (size_t)RB * RN;        // 16384 i32

    // kernel 1: 16384 ROIs / 16 per wave = 1024 waves = 128 blocks x 256
    roi_decode_kernel<<<128, 256, 0, stream>>>(
        logits, deltas, props, sizes, wsBoxes, wsScores, wsLabels, wsValid);

    // kernel 2: one 1024-thread block per image
    roi_nms_kernel<<<RB, RN, 0, stream>>>(
        wsBoxes, wsScores, wsLabels, wsValid, out);
}
